// VanillaRNN_59828894433393
// MI455X (gfx1250) — compile-verified, matrix operand order checked by reading, not run on hardware
//
#include <hip/hip_runtime.h>
#include <math.h>

typedef __attribute__((ext_vector_type(16))) __bf16 v16bf;
typedef __attribute__((ext_vector_type(8)))  __bf16 v8bf;
typedef __attribute__((ext_vector_type(8)))  float  v8f;

static constexpr int SEQ = 512, IN = 512, HID = 1024, OUTD = 512, BATCH = 64;
static constexpr int NBLK = 32;   // persistent-kernel grid size (all resident)

// ---------- scalar conversion helpers ----------

__device__ __forceinline__ __bf16 f2bf(float f) {           // RNE, bit trick
  unsigned u = __builtin_bit_cast(unsigned, f);
  unsigned r = u + 0x7FFFu + ((u >> 16) & 1u);
  unsigned short h = (unsigned short)(r >> 16);
  return __builtin_bit_cast(__bf16, h);
}
__device__ __forceinline__ float bf2f(__bf16 h) {
  unsigned v = ((unsigned)__builtin_bit_cast(unsigned short, h)) << 16;
  return __builtin_bit_cast(float, v);
}

// Lane's 16 bf16 elements of an A/B fragment (16x32 K-major, ISA 7.12.2):
// elements 0..7 = K[k0+8*half .. +7], elements 8..15 = K[k0+16+8*half .. +7]
__device__ __forceinline__ v16bf frag_bf16(const __bf16* __restrict__ row, int k0, int half) {
  const v8bf lo = *(const v8bf*)(row + k0 + half * 8);
  const v8bf hi = *(const v8bf*)(row + k0 + 16 + half * 8);
  v16bf r;
#pragma unroll
  for (int i = 0; i < 8; ++i) { r[i] = lo[i]; r[i + 8] = hi[i]; }
  return r;
}

// ---------- utility kernels ----------

__global__ void f32_to_bf16_vec8(const float* __restrict__ src, __bf16* __restrict__ dst, int n8) {
  int i = blockIdx.x * blockDim.x + threadIdx.x;
  if (i >= n8) return;
  const float4 a = ((const float4*)src)[2 * i];
  const float4 b = ((const float4*)src)[2 * i + 1];
  v8bf o;
  o[0] = f2bf(a.x); o[1] = f2bf(a.y); o[2] = f2bf(a.z); o[3] = f2bf(a.w);
  o[4] = f2bf(b.x); o[5] = f2bf(b.y); o[6] = f2bf(b.z); o[7] = f2bf(b.w);
  ((v8bf*)dst)[i] = o;
}

__global__ void zero_u32_kernel(unsigned* __restrict__ dst, int n) {
  int i = blockIdx.x * blockDim.x + threadIdx.x;
  if (i < n) dst[i] = 0u;
}

// ---------- phase 1: xu = bf16(x @ Wu^T + bu)  (M=32768, N=1024, K=512) ----------
// One wave: 16x128 strip, 2-stage software pipeline over K. waves_per_eu(2)
// opens the VGPR budget so both fragment-buffer generations stay live and
// loads overlap WMMAs instead of s_wait_loadcnt 0 serialization.
__global__ void __launch_bounds__(128)
__attribute__((amdgpu_waves_per_eu(2)))
xu_gemm_kernel(const __bf16* __restrict__ x, const __bf16* __restrict__ wu,
               const float* __restrict__ bu, __bf16* __restrict__ xu) {
  const int wave = (blockIdx.x * blockDim.x + threadIdx.x) >> 5;
  const int lane = threadIdx.x & 31;
  const int nGroup = wave & 7;         // 8 groups of 128 columns
  const int mTile  = wave >> 3;        // 2048 row tiles
  const int m0 = mTile * 16;
  const int n0 = nGroup * 128;
  const int half = lane >> 4, lr = lane & 15;

  const __bf16* arow = x + (size_t)(m0 + lr) * IN;
  v8f acc[8] = {};

  v16bf aBuf[2];
  v16bf bBuf[2][8];
  aBuf[0] = frag_bf16(arow, 0, half);
#pragma unroll
  for (int j = 0; j < 8; ++j)
    bBuf[0][j] = frag_bf16(wu + (size_t)(n0 + j * 16 + lr) * IN, 0, half);

#pragma unroll
  for (int k0 = 0; k0 < IN; k0 += 32) {
    const int cur = (k0 >> 5) & 1, nxt = cur ^ 1;
    if (k0 + 32 < IN) {                       // prefetch next K-group first
      aBuf[nxt] = frag_bf16(arow, k0 + 32, half);
#pragma unroll
      for (int j = 0; j < 8; ++j)
        bBuf[nxt][j] = frag_bf16(wu + (size_t)(n0 + j * 16 + lr) * IN, k0 + 32, half);
    }
#pragma unroll
    for (int j = 0; j < 8; ++j)
      acc[j] = __builtin_amdgcn_wmma_f32_16x16x32_bf16(
          false, aBuf[cur], false, bBuf[cur][j], (short)0, acc[j], false, false);
  }

#pragma unroll
  for (int j = 0; j < 8; ++j) {
#pragma unroll
    for (int r = 0; r < 8; ++r) {
      int row = m0 + r + half * 8;
      int col = n0 + j * 16 + lr;
      xu[(size_t)row * HID + col] = f2bf(acc[j][r] + bu[col]);
    }
  }
}

// ---------- device-wide sense-reversing barrier for the persistent kernel ----------
__device__ __forceinline__ void grid_barrier(unsigned* count, unsigned* gen) {
  __syncthreads();
  if (threadIdx.x == 0) {
    __threadfence();   // make this block's state stores device-visible
    unsigned g = __hip_atomic_load(gen, __ATOMIC_ACQUIRE, __HIP_MEMORY_SCOPE_AGENT);
    unsigned arrived = __hip_atomic_fetch_add(count, 1u, __ATOMIC_ACQ_REL, __HIP_MEMORY_SCOPE_AGENT);
    if (arrived == NBLK - 1) {
      __hip_atomic_store(count, 0u, __ATOMIC_RELAXED, __HIP_MEMORY_SCOPE_AGENT);
      __hip_atomic_fetch_add(gen, 1u, __ATOMIC_ACQ_REL, __HIP_MEMORY_SCOPE_AGENT);
    } else {
      while (__hip_atomic_load(gen, __ATOMIC_ACQUIRE, __HIP_MEMORY_SCOPE_AGENT) == g)
        __builtin_amdgcn_s_sleep(2);
    }
  }
  __syncthreads();
}

// ---------- phase 2: all 512 recurrent steps in ONE persistent kernel ----------
// 32 blocks x 8 waves (1 block/WGP, no wave-level latency hiding -> request
// waves_per_eu(1) for a maximal register budget). Block b owns Ww rows
// [32b,32b+32) in LDS, XOR-swizzled. K=1024 runs on 4 independent accumulator
// chains; A fragments double-buffered one K-group ahead; xu seed loads issue
// at step start and are consumed only in the epilogue.
__global__ void __launch_bounds__(256)
__attribute__((amdgpu_waves_per_eu(1)))
rnn_persistent_kernel(const __bf16* __restrict__ ww, const float* __restrict__ bw,
                      const __bf16* __restrict__ xu, __bf16* __restrict__ s0,
                      __bf16* __restrict__ s1, unsigned* __restrict__ sync) {
  __shared__ v8bf tile[32][128];               // 32 rows x 1024 bf16 = 64 KB

  // --- stage this block's Ww rows into LDS (once) ---
  const __bf16* wsrc = ww + (size_t)(blockIdx.x * 32) * HID;
  for (int c = threadIdx.x; c < 32 * 128; c += blockDim.x) {
    int row = c >> 7, col = c & 127;
    tile[row][col ^ (row & 15)] = ((const v8bf*)wsrc)[(row << 7) | col];
  }
  __syncthreads();

  const int wave = threadIdx.x >> 5, lane = threadIdx.x & 31;
  const int mTile = wave & 3, nSub = wave >> 2;
  const int m0 = mTile * 16;
  const int n0 = blockIdx.x * 32 + nSub * 16;
  const int half = lane >> 4, lr = lane & 15;
  const int lrow = nSub * 16 + lr;             // local LDS row for B fragment
  const int lswz = lrow & 15;
  const float bwv = bw[n0 + lr];

  for (int t = 0; t < SEQ; ++t) {
    const __bf16* sPrev = (t & 1) ? s1 : s0;
    __bf16*       sNext = (t & 1) ? s0 : s1;

    // issue xu seed loads early; consumed only in the epilogue
    float xv[8];
#pragma unroll
    for (int r = 0; r < 8; ++r) {
      int b = m0 + r + half * 8;
      xv[r] = bf2f(xu[((size_t)b * SEQ + t) * HID + (n0 + lr)]);
    }
    // warm next timestep's seed line while this step computes
    if (t + 1 < SEQ)
      __builtin_prefetch(&xu[((size_t)(m0 + lr) * SEQ + (t + 1)) * HID + n0], 0, 3);

    v8f acc[4] = {};
    const __bf16* arow = sPrev + (size_t)(m0 + lr) * HID;

    v16bf aBuf[2][4];
#pragma unroll
    for (int c = 0; c < 4; ++c) aBuf[0][c] = frag_bf16(arow, c * 32, half);

#pragma unroll
    for (int k0 = 0; k0 < HID; k0 += 128) {
      const int cur = (k0 >> 7) & 1, nxt = cur ^ 1;
      if (k0 + 128 < HID) {                    // prefetch next A K-group first
#pragma unroll
        for (int c = 0; c < 4; ++c)
          aBuf[nxt][c] = frag_bf16(arow, k0 + 128 + c * 32, half);
      }
#pragma unroll
      for (int c = 0; c < 4; ++c) {
        const int kc = k0 + c * 32;
        const int c0 = (kc >> 3) + half;       // 16B-chunk index of low half
        v8bf lo = tile[lrow][c0 ^ lswz];
        v8bf hi = tile[lrow][(c0 + 2) ^ lswz];
        v16bf b;
#pragma unroll
        for (int i = 0; i < 8; ++i) { b[i] = lo[i]; b[i + 8] = hi[i]; }
        acc[c] = __builtin_amdgcn_wmma_f32_16x16x32_bf16(
            false, aBuf[cur][c], false, b, (short)0, acc[c], false, false);
      }
    }

#pragma unroll
    for (int r = 0; r < 8; ++r) {
      int b = m0 + r + half * 8;
      float v = (acc[0][r] + acc[1][r]) + (acc[2][r] + acc[3][r]) + xv[r];
      sNext[(size_t)b * HID + (n0 + lr)] = f2bf(tanhf(v + bwv));
    }

    grid_barrier(sync, sync + 1);              // all 32 blocks step together
  }
}

// ---------- phase 3: out = s_final @ Wv^T + bv  (M=64, N=512, K=1024) ----------
__global__ void __launch_bounds__(128)
__attribute__((amdgpu_waves_per_eu(1)))
out_gemm_kernel(const __bf16* __restrict__ s, const __bf16* __restrict__ wv,
                const float* __restrict__ bv, float* __restrict__ out) {
  const int wave = (blockIdx.x * blockDim.x + threadIdx.x) >> 5;  // 0..127
  const int lane = threadIdx.x & 31;
  const int nTile = wave & 31, mTile = wave >> 5;
  const int m0 = mTile * 16, n0 = nTile * 16;
  const int half = lane >> 4, lr = lane & 15;

  const __bf16* arow = s  + (size_t)(m0 + lr) * HID;
  const __bf16* brow = wv + (size_t)(n0 + lr) * HID;
  v8f acc[4] = {};
  for (int k0 = 0; k0 < HID; k0 += 128) {
#pragma unroll
    for (int c = 0; c < 4; ++c) {
      const int kc = k0 + c * 32;
      v16bf a = frag_bf16(arow, kc, half);
      v16bf b = frag_bf16(brow, kc, half);
      acc[c] = __builtin_amdgcn_wmma_f32_16x16x32_bf16(
          false, a, false, b, (short)0, acc[c], false, false);
    }
  }
#pragma unroll
  for (int r = 0; r < 8; ++r) {
    float v = (acc[0][r] + acc[1][r]) + (acc[2][r] + acc[3][r]);
    out[(size_t)(m0 + r + half * 8) * OUTD + (n0 + lr)] = v + bv[n0 + lr];
  }
}

// ---------- host orchestration ----------

extern "C" void kernel_launch(void* const* d_in, const int* in_sizes, int n_in,
                              void* d_out, int out_size, void* d_ws, size_t ws_size,
                              hipStream_t stream) {
  (void)in_sizes; (void)n_in; (void)out_size; (void)ws_size;
  const float* x  = (const float*)d_in[0];
  const float* Wu = (const float*)d_in[1];
  const float* bu = (const float*)d_in[2];
  const float* Ww = (const float*)d_in[3];
  const float* bw = (const float*)d_in[4];
  const float* Wv = (const float*)d_in[5];
  const float* bv = (const float*)d_in[6];
  float* out = (float*)d_out;

  // workspace layout (~101 MB total)
  char* ws = (char*)d_ws;
  __bf16*   wu_bf = (__bf16*)(ws);                            // 1 MB
  __bf16*   ww_bf = (__bf16*)(ws + ((size_t)1 << 20));        // 2 MB
  __bf16*   wv_bf = (__bf16*)(ws + ((size_t)3 << 20));        // 1 MB
  __bf16*   s0    = (__bf16*)(ws + ((size_t)4 << 20));        // 128 KB
  __bf16*   s1    = (__bf16*)(ws + ((size_t)4 << 20) + 131072);
  unsigned* sync  = (unsigned*)(ws + ((size_t)4 << 20) + 262144);
  __bf16*   x_bf  = (__bf16*)(ws + ((size_t)5 << 20));        // 32 MB
  __bf16*   xu_bf = (__bf16*)(ws + ((size_t)37 << 20));       // 64 MB

  // 1. one-time bf16 conversions (weights + x); zero state + barrier words
  f32_to_bf16_vec8<<<(HID * IN / 8 + 255) / 256, 256, 0, stream>>>(Wu, wu_bf, HID * IN / 8);
  f32_to_bf16_vec8<<<(HID * HID / 8 + 255) / 256, 256, 0, stream>>>(Ww, ww_bf, HID * HID / 8);
  f32_to_bf16_vec8<<<(OUTD * HID / 8 + 255) / 256, 256, 0, stream>>>(Wv, wv_bf, OUTD * HID / 8);
  f32_to_bf16_vec8<<<(BATCH * SEQ * IN / 8 + 255) / 256, 256, 0, stream>>>(x, x_bf, BATCH * SEQ * IN / 8);
  zero_u32_kernel<<<(BATCH * HID / 2 + 255) / 256, 256, 0, stream>>>((unsigned*)s0, BATCH * HID / 2);
  zero_u32_kernel<<<1, 32, 0, stream>>>(sync, 2);

  // 2. input-projection GEMM: 16384 waves, software-pipelined b128-load + WMMA
  xu_gemm_kernel<<<4096, 128, 0, stream>>>(x_bf, wu_bf, bu, xu_bf);

  // 3. whole recurrence in one persistent kernel (Ww LDS-resident, grid barrier per step)
  rnn_persistent_kernel<<<NBLK, 256, 0, stream>>>(ww_bf, bw, xu_bf, s0, s1, sync);
  // t=511 (odd) wrote s0 -> final state in s0

  // 4. output projection
  out_gemm_kernel<<<32, 128, 0, stream>>>(s0, wv_bf, bv, out);
}
